// EdgePooling_80642305950444
// MI455X (gfx1250) — matrix operand order repaired and checked by reading, not compiled, not run on hardware
//
#include <hip/hip_runtime.h>
#include <hip/hip_bf16.h>

typedef __bf16 v16bf __attribute__((ext_vector_type(16)));
typedef __bf16 v8bf  __attribute__((ext_vector_type(8)));
typedef __bf16 v2bf  __attribute__((ext_vector_type(2)));
typedef float  v8f   __attribute__((ext_vector_type(8)));
typedef float  v4f   __attribute__((ext_vector_type(4)));

#define DIM    128
#define TILE_M 16
#define LDA    144   // padded bf16 row stride: 288 B (16B-aligned rows, staggered banks)

// ds_swizzle group-of-32 XOR patterns: offset = xor<<10 | or<<5 | and(0x1f)
#define SWZ_ADD(p, imm)                                                        \
    (p) += __builtin_bit_cast(float,                                           \
        __builtin_amdgcn_ds_swizzle(__builtin_bit_cast(int, (p)), (imm)))

#if __has_builtin(__builtin_amdgcn_cvt_pk_bf16_f32)
__device__ __forceinline__ v2bf pkbf(float a, float b) {
    return __builtin_amdgcn_cvt_pk_bf16_f32(a, b);
}
#else
__device__ __forceinline__ __bf16 f2bf_(float f) {
    unsigned u = __builtin_bit_cast(unsigned, f);
    u += 0x7FFFu + ((u >> 16) & 1u);          // round-to-nearest-even
    unsigned short h = (unsigned short)(u >> 16);
    return __builtin_bit_cast(__bf16, h);
}
__device__ __forceinline__ v2bf pkbf(float a, float b) {
    v2bf r; r[0] = f2bf_(a); r[1] = f2bf_(b); return r;
}
#endif

__device__ __forceinline__ float bf2f(__bf16 h) {
    unsigned u = ((unsigned)__builtin_bit_cast(unsigned short, h)) << 16;
    return __builtin_bit_cast(float, u);
}

__global__ __launch_bounds__(256) void edgepool_wmma_kernel(
    const float* __restrict__ x,
    const long long* __restrict__ edge_index,   // [2,E] int64
    const float* __restrict__ W_t,              // [128,128] row-major (n,k)
    const float* __restrict__ b_t,              // [128]
    const float* __restrict__ W_s,              // [128]
    const float* __restrict__ b_sp,             // [1]
    float* __restrict__ out_edges,              // [E,128]
    float* __restrict__ out_self,               // [N,128]
    float* __restrict__ out_score,              // [E]
    int E, int N)
{
    __shared__ __bf16 A_lds[TILE_M * LDA];

    const int tid   = threadIdx.x;
    const int wave  = tid >> 5;         // 0..7 -> column tile
    const int lane  = tid & 31;
    const int lhalf = lane >> 4;        // 0 | 1
    const int l16   = lane & 15;
    const int col   = wave * 16 + l16;  // output column owned in C/D layout

    // ---- B fragments (W_t^T as 128x16 bf16, per K-step of 32), loaded once ----
    // B 32x16 k order chosen here (contiguous, same as A): lanes 0-15 hold K=kb..kb+15 of col n, lanes 16-31 K=kb+16..+31.
    // Those K are contiguous in W_t row n -> plain 64B global reads, no transpose.
    v16bf Bfrag[4];
    {
        const float* wrow = W_t + (size_t)col * DIM;
        #pragma unroll
        for (int s = 0; s < 4; ++s) {
            const int kb = s * 32 + lhalf * 16;
            union { v16bf v; v2bf p[8]; } u;
            #pragma unroll
            for (int i = 0; i < 8; ++i)
                u.p[i] = pkbf(wrow[kb + 2 * i], wrow[kb + 2 * i + 1]);
            Bfrag[s] = u.v;
        }
    }
    const float bias = b_t[col];
    const float bs   = b_sp[0];

    // ---- score precompute: v = W_t^T @ W_s (this thread's 8-k chunk) and
    //      c = b_t . W_s + b_s.  score_row = A_row . v + c  (exact rewrite) ----
    const int schunk = (tid & 15) * 8;        // k-chunk owned for score dot
    const int srowS  = tid >> 4;              // row owned for score dot
    float vws[8];
    float cb = bs;
    {
        #pragma unroll
        for (int j = 0; j < 8; ++j) vws[j] = 0.0f;
        for (int n = 0; n < DIM; ++n) {
            const float wsn = W_s[n];
            const float* wr = W_t + (size_t)n * DIM + schunk;
            #pragma unroll
            for (int j = 0; j < 8; ++j) vws[j] = fmaf(wr[j], wsn, vws[j]);
            cb = fmaf(b_t[n], wsn, cb);
        }
    }

    const int nTiles = (E + N) >> 4;    // E and N are multiples of 16

    // staging decomposition: thread handles row tid/16, 8 floats at col (tid%16)*8
    const int srow = tid >> 4;
    const int scol = (tid & 15) * 8;

    for (int tile = blockIdx.x; tile < nTiles; tile += gridDim.x) {
        const int  row0   = tile * TILE_M;
        const bool isEdge = row0 < E;   // uniform across the block

        // ---- stage A = bf16(x[src]+x[dst]) (or bf16(2x)) into LDS ----
        {
            const int r = row0 + srow;
            const float *p0, *p1;
            if (isEdge) {
                const long long s = __builtin_nontemporal_load(edge_index + r);
                const long long d = __builtin_nontemporal_load(edge_index + (size_t)E + r);
                p0 = x + (size_t)s * DIM + scol;
                p1 = x + (size_t)d * DIM + scol;
            } else {
                p0 = x + (size_t)(r - E) * DIM + scol;
                p1 = p0;
            }
            const v4f a0 = *(const v4f*)p0;
            const v4f a1 = *(const v4f*)(p0 + 4);
            const v4f b0 = *(const v4f*)p1;
            const v4f b1 = *(const v4f*)(p1 + 4);
            union { v8bf v; v2bf p[4]; } u;
            u.p[0] = pkbf(a0[0] + b0[0], a0[1] + b0[1]);
            u.p[1] = pkbf(a0[2] + b0[2], a0[3] + b0[3]);
            u.p[2] = pkbf(a1[0] + b1[0], a1[1] + b1[1]);
            u.p[3] = pkbf(a1[2] + b1[2], a1[3] + b1[3]);
            *(v8bf*)&A_lds[srow * LDA + scol] = u.v;   // single ds_store_b128
        }
        __syncthreads();

        // ---- GEMM: D(16x16) = A(16x128) x B(128x16), 4 bf16 WMMA steps ----
        v8f acc = {};
        #pragma unroll
        for (int s = 0; s < 4; ++s) {
            const int kb = s * 32;
            // A 16x32 layout: lane<16 K=kb..kb+15 contiguous;
            //                 lane>=16 K=kb+16..kb+31 (matches B).
            const __bf16* ap = &A_lds[l16 * LDA + kb + lhalf * 16];
            v16bf afr;
            *(v8bf*)&afr       = *(const v8bf*)ap;         // ds_load_b128
            *((v8bf*)&afr + 1) = *(const v8bf*)(ap + 8);   // ds_load_b128
            acc = __builtin_amdgcn_wmma_f32_16x16x32_bf16(
                false, afr, false, Bfrag[s], (short)0, acc, false, false);
        }
        #pragma unroll
        for (int m = 0; m < 8; ++m) acc[m] += bias;   // per-column bias

        // ---- store D (non-temporal: 877MB stream must not thrash L2) ----
        {
            float* obase = isEdge ? (out_edges + (size_t)row0 * DIM)
                                  : (out_self  + (size_t)(row0 - E) * DIM);
            const int m0 = lhalf * 8;
            #pragma unroll
            for (int m = 0; m < 8; ++m)
                __builtin_nontemporal_store(
                    acc[m], obase + (size_t)(m0 + m) * DIM + col);
        }

        // ---- fused score: e = sigmoid(A_row . v + c) ----
        if (isEdge) {
            const v8bf arow = *(const v8bf*)&A_lds[srowS * LDA + schunk];
            float p = 0.0f;
            #pragma unroll
            for (int j = 0; j < 8; ++j) p = fmaf(bf2f(arow[j]), vws[j], p);
            SWZ_ADD(p, 0x041F);   // xor 1
            SWZ_ADD(p, 0x081F);   // xor 2
            SWZ_ADD(p, 0x101F);   // xor 4
            SWZ_ADD(p, 0x201F);   // xor 8  -> sum over the 16-lane chunk group
            if ((tid & 15) == 0) {
                const float vsc = p + cb;
                __builtin_nontemporal_store(
                    1.0f / (1.0f + __expf(-vsc)), out_score + row0 + srowS);
            }
        }
        __syncthreads();   // protect A_lds for next iteration
    }
}

extern "C" void kernel_launch(void* const* d_in, const int* in_sizes, int n_in,
                              void* d_out, int out_size, void* d_ws, size_t ws_size,
                              hipStream_t stream) {
    const float*     x   = (const float*)d_in[0];
    const long long* ei  = (const long long*)d_in[1];
    // d_in[2] = batch (unused by the outputs)
    const float*     W_t = (const float*)d_in[3];
    const float*     b_t = (const float*)d_in[4];
    const float*     W_s = (const float*)d_in[5];
    const float*     b_s = (const float*)d_in[6];

    const int N = in_sizes[0] / DIM;
    const int E = in_sizes[1] / 2;

    float* out       = (float*)d_out;
    float* out_edges = out;                                  // [E,128]
    float* out_self  = out + (size_t)E * DIM;                // [N,128]
    float* out_score = out + ((size_t)E + (size_t)N) * DIM;  // [E]

    dim3 grid(2048), block(256);
    edgepool_wmma_kernel<<<grid, block, 0, stream>>>(
        x, ei, W_t, b_t, W_s, b_s, out_edges, out_self, out_score, E, N);
}
